// DecoderLayer_2336462209132
// MI455X (gfx1250) — compile-verified, matrix operand order checked by reading, not run on hardware
//
#include <hip/hip_runtime.h>

// ---------------------------------------------------------------------------
// Problem constants (from reference): D=1024, H=16, DK=64, DFF=4096, B=2,
// T=S=2048.  M = B*T = 4096 rows for every activation GEMM.
// ---------------------------------------------------------------------------
#define DMODEL 1024
#define NHEAD  16
#define DHEAD  64
#define DFFN   4096
#define NB     2
#define TSEQ   2048
#define SSEQ   2048
#define MROWS  (NB * TSEQ)   // 4096

typedef __attribute__((ext_vector_type(16))) __bf16 v16bf;
typedef __attribute__((ext_vector_type(8)))  __bf16 v8bf;
typedef __attribute__((ext_vector_type(8)))  float  v8f;

// ---------------------------------------------------------------------------
// Helpers
// ---------------------------------------------------------------------------
__device__ __forceinline__ __bf16 f2bf(float f) {
  unsigned int u = __float_as_uint(f);
  u += 0x7FFFu + ((u >> 16) & 1u);          // round-to-nearest-even
  unsigned short s = (unsigned short)(u >> 16);
  __bf16 r;
  __builtin_memcpy(&r, &s, 2);
  return r;
}

// Load one WMMA 16-lane fragment row following the CDNA5 per-lane K-chunk
// layout: lanes 0-15 hold K {hi8=0: 0..7,16..23}, lanes 16-31 hold {8..15,24..31}.
__device__ __forceinline__ v16bf load_frag(const __bf16* row, int hi8) {
  v8bf a = *(const v8bf*)(row + hi8);
  v8bf b = *(const v8bf*)(row + hi8 + 16);
  return __builtin_shufflevector(a, b, 0,1,2,3,4,5,6,7,8,9,10,11,12,13,14,15);
}

__device__ __forceinline__ v8f wmma_bf16(v16bf a, v16bf b, v8f c) {
  return __builtin_amdgcn_wmma_f32_16x16x32_bf16(
      /*neg_a=*/false, a, /*neg_b=*/false, b,
      /*c_mod=*/(short)0, c, /*reuse_a=*/false, /*reuse_b=*/false);
}

// CDNA5 async global->LDS copy (16B per lane), tracked by ASYNCcnt.
// lds_addr = byte offset within the workgroup LDS allocation (low 32 bits of
// the generic pointer), gptr = per-lane 64-bit global address.
__device__ __forceinline__ void async_copy_b128(unsigned int lds_addr,
                                                const void* gptr) {
  asm volatile("global_load_async_to_lds_b128 %0, %1, off"
               :: "v"(lds_addr), "v"(gptr)
               : "memory");
}
__device__ __forceinline__ void wait_asynccnt0() {
  asm volatile("s_wait_asynccnt 0x0" ::: "memory");
}

// ---------------------------------------------------------------------------
// fp32 -> bf16 conversion (grid-stride)
// ---------------------------------------------------------------------------
__global__ __launch_bounds__(256) void cvt_kernel(const float* __restrict__ in,
                                                  __bf16* __restrict__ out, int n) {
  int i = blockIdx.x * 256 + threadIdx.x;
  const int stride = gridDim.x * 256;
  for (; i < n; i += stride) out[i] = f2bf(in[i]);
}

// ---------------------------------------------------------------------------
// Tiled GEMM: out[M,N] = A[M,K] @ W[N,K]^T + bias[N]   (torch Linear semantics)
// Block = 256 threads = 8 waves; 128x128 output tile; wave tile 32x64.
// K is staged 32 wide through double-buffered LDS using async global->LDS
// copies so the next tile streams in while WMMAs consume the current one.
// ---------------------------------------------------------------------------
template<bool RELU, bool OUT_BF16>
__global__ __launch_bounds__(256) void gemm_kernel(
    const __bf16* __restrict__ A, const __bf16* __restrict__ W,
    const float* __restrict__ bias, void* __restrict__ outp,
    int M, int N, int K)
{
  __shared__ __bf16 As[2][128 * 32];
  __shared__ __bf16 Ws[2][128 * 32];

  const int tid  = threadIdx.x;
  const int lane = tid & 31;
  const int wave = tid >> 5;     // 0..7
  const int wm   = wave >> 1;    // 0..3 -> 32-row strip
  const int wn   = wave & 1;     // 0..1 -> 64-col strip
  const int bm   = blockIdx.x * 128;
  const int bn   = blockIdx.y * 128;
  const int l15  = lane & 15;
  const int hi8  = (lane >> 4) << 3;

  const v8f vzero = {0.f,0.f,0.f,0.f,0.f,0.f,0.f,0.f};
  v8f acc[2][4];
#pragma unroll
  for (int i = 0; i < 2; ++i)
#pragma unroll
    for (int j = 0; j < 4; ++j) acc[i][j] = vzero;

  const int ldrow = tid >> 1;          // 0..127
  const int ldcol = (tid & 1) * 16;    // 0 or 16  (32 contiguous bytes/thread)
  const __bf16* aptr = A + (size_t)(bm + ldrow) * K + ldcol;
  const __bf16* wptr = W + (size_t)(bn + ldrow) * K + ldcol;
  const unsigned int lA0 = (unsigned int)(uintptr_t)(&As[0][ldrow * 32 + ldcol]);
  const unsigned int lA1 = (unsigned int)(uintptr_t)(&As[1][ldrow * 32 + ldcol]);
  const unsigned int lW0 = (unsigned int)(uintptr_t)(&Ws[0][ldrow * 32 + ldcol]);
  const unsigned int lW1 = (unsigned int)(uintptr_t)(&Ws[1][ldrow * 32 + ldcol]);

  auto stage = [&](int buf, int k0) {
    const unsigned int la = buf ? lA1 : lA0;
    const unsigned int lw = buf ? lW1 : lW0;
    async_copy_b128(la,      aptr + k0);
    async_copy_b128(la + 16, aptr + k0 + 8);
    async_copy_b128(lw,      wptr + k0);
    async_copy_b128(lw + 16, wptr + k0 + 8);
  };

  stage(0, 0);
  int buf = 0;
  for (int k0 = 0; k0 < K; k0 += 32) {
    wait_asynccnt0();     // our async fills for `buf` have landed in LDS
    __syncthreads();      // ... and everyone else's too; prev reads done
    if (k0 + 32 < K) stage(buf ^ 1, k0 + 32);   // prefetch next tile

    v16bf af[2], bfr[4];
#pragma unroll
    for (int i = 0; i < 2; ++i)
      af[i] = load_frag(&As[buf][(wm * 32 + i * 16 + l15) * 32], hi8);
#pragma unroll
    for (int j = 0; j < 4; ++j)
      bfr[j] = load_frag(&Ws[buf][(wn * 64 + j * 16 + l15) * 32], hi8);
#pragma unroll
    for (int i = 0; i < 2; ++i)
#pragma unroll
      for (int j = 0; j < 4; ++j)
        acc[i][j] = wmma_bf16(af[i], bfr[j], acc[i][j]);
    buf ^= 1;
  }

  float*  outf = (float*)outp;
  __bf16* outb = (__bf16*)outp;
#pragma unroll
  for (int i = 0; i < 2; ++i) {
    const int mrow = bm + wm * 32 + i * 16 + hi8;  // C layout: row r+8*(lane>>4)
#pragma unroll
    for (int j = 0; j < 4; ++j) {
      const int ncol = bn + wn * 64 + j * 16 + l15;
      const float bv = bias[ncol];
#pragma unroll
      for (int r = 0; r < 8; ++r) {
        float v = acc[i][j][r] + bv;
        if (RELU) v = fmaxf(v, 0.0f);
        const size_t idx = (size_t)(mrow + r) * N + ncol;
        if (OUT_BF16) outb[idx] = f2bf(v); else outf[idx] = v;
      }
    }
  }
}

// ---------------------------------------------------------------------------
// Transpose V ([b][s][h][d] bf16) -> Vt ([b][h][d][s] bf16) so V columns are
// contiguous for the PV B-operand.
// ---------------------------------------------------------------------------
__global__ __launch_bounds__(256) void transpose_v_kernel(
    const __bf16* __restrict__ V, __bf16* __restrict__ Vt, int Skv)
{
  __shared__ __bf16 tile[32][72];    // 32 s  x 64 d (+pad)
  const int b  = blockIdx.z;
  const int h  = blockIdx.y;
  const int s0 = blockIdx.x * 32;
  const int tid = threadIdx.x;
  {
    const int sl = tid >> 3;                 // 0..31
    const int dp = (tid & 7) * 8;            // 0..56
    const __bf16* src = V + (size_t)(b * Skv + s0 + sl) * DMODEL + h * DHEAD + dp;
    v8bf val = *(const v8bf*)src;
#pragma unroll
    for (int i = 0; i < 8; ++i) tile[sl][dp + i] = val[i];
  }
  __syncthreads();
  {
    const int dl = tid >> 2;                 // 0..63
    const int sp = (tid & 3) * 8;            // 0..24
    v8bf o;
#pragma unroll
    for (int i = 0; i < 8; ++i) o[i] = tile[sp + i][dl];
    __bf16* dst = Vt + (size_t)((b * NHEAD + h) * DHEAD + dl) * Skv + s0 + sp;
    *(v8bf*)dst = o;
  }
}

// ---------------------------------------------------------------------------
// Flash attention. Block = 128 threads = 4 waves, each wave owns 16 q rows.
// Per 32-column S step: 4 WMMAs for Q@K^T (K-dim 64), online softmax with
// cross-lane reductions, P -> LDS -> A-fragment, 4 WMMAs for P@V.
// ---------------------------------------------------------------------------
__global__ __launch_bounds__(128) void attn_kernel(
    const __bf16* __restrict__ Qb,   // [B*Tq, DMODEL] bf16
    const __bf16* __restrict__ Kb,   // [B*Skv, DMODEL] bf16
    const __bf16* __restrict__ Vt,   // [B*H*DHEAD, Skv] bf16
    const int*    __restrict__ mask, // [B, Tq, Skv]
    float*        __restrict__ ctx,  // [B*Tq, DMODEL] f32
    int Tq, int Skv)
{
  __shared__ __bf16 plds[4][16 * 32];
  const int lane = threadIdx.x & 31;
  const int wave = threadIdx.x >> 5;
  const int l15  = lane & 15;
  const int hi8  = (lane >> 4) << 3;
  const int h    = blockIdx.y;
  const int b    = blockIdx.z;
  const int q0   = blockIdx.x * 64 + wave * 16;

  const __bf16* qrow = Qb + (size_t)(b * Tq + q0 + l15) * DMODEL + h * DHEAD;
  const v16bf qf0 = load_frag(qrow, hi8);        // K-dim 0..31 of head
  const v16bf qf1 = load_frag(qrow + 32, hi8);   // K-dim 32..63

  const __bf16* kbase  = Kb + (size_t)b * Skv * DMODEL + h * DHEAD;
  const __bf16* vtbase = Vt + (size_t)((b * NHEAD + h) * DHEAD) * Skv;
  const int*    mbase  = mask + (size_t)b * Tq * Skv;

  const v8f vzero = {0.f,0.f,0.f,0.f,0.f,0.f,0.f,0.f};
  v8f o[4];
#pragma unroll
  for (int j = 0; j < 4; ++j) o[j] = vzero;
  float mrow[8], lrow[8];
#pragma unroll
  for (int r = 0; r < 8; ++r) { mrow[r] = -1e30f; lrow[r] = 0.f; }

  __bf16* pl = plds[wave];

  for (int s0 = 0; s0 < Skv; s0 += 32) {
    v8f sc0 = vzero, sc1 = vzero;
    {
      const __bf16* krow0 = kbase + (size_t)(s0 + l15) * DMODEL;
      const __bf16* krow1 = kbase + (size_t)(s0 + 16 + l15) * DMODEL;
      sc0 = wmma_bf16(qf0, load_frag(krow0, hi8), sc0);
      sc0 = wmma_bf16(qf1, load_frag(krow0 + 32, hi8), sc0);
      sc1 = wmma_bf16(qf0, load_frag(krow1, hi8), sc1);
      sc1 = wmma_bf16(qf1, load_frag(krow1 + 32, hi8), sc1);
    }
    // scale + mask + local row max (C layout: element r -> row q0+r+hi8,
    // col l15 (tile0) / 16+l15 (tile1))
    float nmax[8];
#pragma unroll
    for (int r = 0; r < 8; ++r) {
      float v0 = sc0[r] * 0.125f;   // 1/sqrt(64)
      float v1 = sc1[r] * 0.125f;
      const int* mrp = mbase + (size_t)(q0 + r + hi8) * Skv + s0;
      if (mrp[l15] == 0)      v0 = -1e9f;
      if (mrp[16 + l15] == 0) v1 = -1e9f;
      sc0[r] = v0; sc1[r] = v1;
      nmax[r] = fmaxf(v0, v1);
    }
#pragma unroll
    for (int off = 1; off < 16; off <<= 1)
#pragma unroll
      for (int r = 0; r < 8; ++r)
        nmax[r] = fmaxf(nmax[r], __shfl_xor(nmax[r], off, 32));

    float scl[8], rsum[8];
#pragma unroll
    for (int r = 0; r < 8; ++r) {
      const float mn = fmaxf(mrow[r], nmax[r]);
      scl[r] = __expf(mrow[r] - mn);
      mrow[r] = mn;
      const float p0 = __expf(sc0[r] - mn);
      const float p1 = __expf(sc1[r] - mn);
      sc0[r] = p0; sc1[r] = p1;
      rsum[r] = p0 + p1;
    }
#pragma unroll
    for (int off = 1; off < 16; off <<= 1)
#pragma unroll
      for (int r = 0; r < 8; ++r)
        rsum[r] += __shfl_xor(rsum[r], off, 32);
#pragma unroll
    for (int r = 0; r < 8; ++r) lrow[r] = lrow[r] * scl[r] + rsum[r];
#pragma unroll
    for (int j = 0; j < 4; ++j)
#pragma unroll
      for (int r = 0; r < 8; ++r) o[j][r] *= scl[r];

    // P (C layout, f32) -> LDS row-major 16x32 bf16
#pragma unroll
    for (int r = 0; r < 8; ++r) {
      pl[(r + hi8) * 32 + l15]      = f2bf(sc0[r]);
      pl[(r + hi8) * 32 + 16 + l15] = f2bf(sc1[r]);
    }
    __syncthreads();
    const v16bf pf = load_frag(pl + l15 * 32, hi8);   // A-fragment of P
#pragma unroll
    for (int j = 0; j < 4; ++j) {
      const __bf16* vrow = vtbase + (size_t)(j * 16 + l15) * Skv + s0;
      o[j] = wmma_bf16(pf, load_frag(vrow, hi8), o[j]);
    }
    __syncthreads();
  }

#pragma unroll
  for (int r = 0; r < 8; ++r) lrow[r] = 1.0f / lrow[r];
#pragma unroll
  for (int j = 0; j < 4; ++j)
#pragma unroll
    for (int r = 0; r < 8; ++r) {
      const size_t idx = (size_t)(b * Tq + q0 + r + hi8) * DMODEL + h * DHEAD + j * 16 + l15;
      ctx[idx] = o[j][r] * lrow[r];
    }
}

// ---------------------------------------------------------------------------
// Residual add + LayerNorm; writes f32 stream + bf16 copy for next GEMM.
// One block per row of 1024. (x/yout may alias for in-place update.)
// ---------------------------------------------------------------------------
__global__ __launch_bounds__(256) void ln_kernel(
    const float* x, const float* __restrict__ sub,
    const float* __restrict__ g, const float* __restrict__ bta,
    float* yout, __bf16* __restrict__ ybf)
{
  __shared__ float s1[256], s2[256];
  const size_t base = (size_t)blockIdx.x * DMODEL;
  float v[4];
  float lsum = 0.f, lsq = 0.f;
#pragma unroll
  for (int k = 0; k < 4; ++k) {
    const int i = threadIdx.x + k * 256;
    v[k] = x[base + i] + sub[base + i];
    lsum += v[k];
    lsq  += v[k] * v[k];
  }
  s1[threadIdx.x] = lsum; s2[threadIdx.x] = lsq;
  __syncthreads();
  for (int off = 128; off > 0; off >>= 1) {
    if (threadIdx.x < off) {
      s1[threadIdx.x] += s1[threadIdx.x + off];
      s2[threadIdx.x] += s2[threadIdx.x + off];
    }
    __syncthreads();
  }
  const float mean = s1[0] * (1.0f / DMODEL);
  const float var  = s2[0] * (1.0f / DMODEL) - mean * mean;
  const float rs   = rsqrtf(var + 1e-5f);
#pragma unroll
  for (int k = 0; k < 4; ++k) {
    const int i = threadIdx.x + k * 256;
    const float o = (v[k] - mean) * rs * g[i] + bta[i];
    yout[base + i] = o;
    ybf[base + i]  = f2bf(o);
  }
}

// ---------------------------------------------------------------------------
// Host orchestration
// ---------------------------------------------------------------------------
extern "C" void kernel_launch(void* const* d_in, const int* in_sizes, int n_in,
                              void* d_out, int out_size, void* d_ws, size_t ws_size,
                              hipStream_t stream) {
  (void)in_sizes; (void)n_in; (void)out_size; (void)ws_size;

  const float* x        = (const float*)d_in[0];
  const float* enc      = (const float*)d_in[1];
  const int*   src_mask = (const int*)d_in[2];
  const int*   tgt_mask = (const int*)d_in[3];
  const float* sa_Wq = (const float*)d_in[4],  *sa_bq = (const float*)d_in[5];
  const float* sa_Wk = (const float*)d_in[6],  *sa_bk = (const float*)d_in[7];
  const float* sa_Wv = (const float*)d_in[8],  *sa_bv = (const float*)d_in[9];
  const float* sa_Wo = (const float*)d_in[10], *sa_bo = (const float*)d_in[11];
  const float* ca_Wq = (const float*)d_in[12], *ca_bq = (const float*)d_in[13];
  const float* ca_Wk = (const float*)d_in[14], *ca_bk = (const float*)d_in[15];
  const float* ca_Wv = (const float*)d_in[16], *ca_bv = (const float*)d_in[17];
  const float* ca_Wo = (const float*)d_in[18], *ca_bo = (const float*)d_in[19];
  const float* ff_W1 = (const float*)d_in[20], *ff_b1 = (const float*)d_in[21];
  const float* ff_W2 = (const float*)d_in[22], *ff_b2 = (const float*)d_in[23];
  const float* ln1_g = (const float*)d_in[24], *ln1_b = (const float*)d_in[25];
  const float* ln2_g = (const float*)d_in[26], *ln2_b = (const float*)d_in[27];
  const float* ln3_g = (const float*)d_in[28], *ln3_b = (const float*)d_in[29];

  char* ws = (char*)d_ws;
  const size_t MB = 1024ull * 1024ull;
  // bf16 weights
  __bf16* saWq_bf = (__bf16*)(ws + 0 * MB);
  __bf16* saWk_bf = (__bf16*)(ws + 2 * MB);
  __bf16* saWv_bf = (__bf16*)(ws + 4 * MB);
  __bf16* saWo_bf = (__bf16*)(ws + 6 * MB);
  __bf16* caWq_bf = (__bf16*)(ws + 8 * MB);
  __bf16* caWk_bf = (__bf16*)(ws + 10 * MB);
  __bf16* caWv_bf = (__bf16*)(ws + 12 * MB);
  __bf16* caWo_bf = (__bf16*)(ws + 14 * MB);
  __bf16* W1_bf   = (__bf16*)(ws + 16 * MB);   // 8 MB
  __bf16* W2_bf   = (__bf16*)(ws + 24 * MB);   // 8 MB
  // activations
  __bf16* x_bf    = (__bf16*)(ws + 32 * MB);
  __bf16* enc_bf  = (__bf16*)(ws + 40 * MB);
  __bf16* Qb      = (__bf16*)(ws + 48 * MB);
  __bf16* Kb      = (__bf16*)(ws + 56 * MB);
  __bf16* Vb      = (__bf16*)(ws + 64 * MB);
  __bf16* Vt      = (__bf16*)(ws + 72 * MB);
  float*  ctx     = (float*) (ws + 80 * MB);   // 16 MB
  float*  proj    = (float*) (ws + 96 * MB);   // 16 MB
  float*  yres    = (float*) (ws + 112 * MB);  // 16 MB
  __bf16* y_bf    = (__bf16*)(ws + 128 * MB);  // 8 MB
  __bf16* hidden  = (__bf16*)(ws + 136 * MB);  // 32 MB
  __bf16* ctx_bf  = (__bf16*)(ws + 168 * MB);  // 8 MB

  auto cvt = [&](const float* in, __bf16* out, int n) {
    cvt_kernel<<<dim3(2048), dim3(256), 0, stream>>>(in, out, n);
  };
  auto gemmF = [&](const __bf16* A, const __bf16* W, const float* b, float* out,
                   int M, int N, int K) {
    gemm_kernel<false, false><<<dim3(M / 128, N / 128), 256, 0, stream>>>(
        A, W, b, (void*)out, M, N, K);
  };
  auto gemmB = [&](const __bf16* A, const __bf16* W, const float* b, __bf16* out,
                   int M, int N, int K) {
    gemm_kernel<false, true><<<dim3(M / 128, N / 128), 256, 0, stream>>>(
        A, W, b, (void*)out, M, N, K);
  };

  // --- precision staging ---
  cvt(sa_Wq, saWq_bf, DMODEL * DMODEL);
  cvt(sa_Wk, saWk_bf, DMODEL * DMODEL);
  cvt(sa_Wv, saWv_bf, DMODEL * DMODEL);
  cvt(sa_Wo, saWo_bf, DMODEL * DMODEL);
  cvt(ca_Wq, caWq_bf, DMODEL * DMODEL);
  cvt(ca_Wk, caWk_bf, DMODEL * DMODEL);
  cvt(ca_Wv, caWv_bf, DMODEL * DMODEL);
  cvt(ca_Wo, caWo_bf, DMODEL * DMODEL);
  cvt(ff_W1, W1_bf, DFFN * DMODEL);
  cvt(ff_W2, W2_bf, DMODEL * DFFN);
  cvt(x,   x_bf,   MROWS * DMODEL);
  cvt(enc, enc_bf, MROWS * DMODEL);

  // --- self attention ---
  gemmB(x_bf, saWq_bf, sa_bq, Qb, MROWS, DMODEL, DMODEL);
  gemmB(x_bf, saWk_bf, sa_bk, Kb, MROWS, DMODEL, DMODEL);
  gemmB(x_bf, saWv_bf, sa_bv, Vb, MROWS, DMODEL, DMODEL);
  transpose_v_kernel<<<dim3(TSEQ / 32, NHEAD, NB), 256, 0, stream>>>(Vb, Vt, TSEQ);
  attn_kernel<<<dim3(TSEQ / 64, NHEAD, NB), 128, 0, stream>>>(
      Qb, Kb, Vt, tgt_mask, ctx, TSEQ, TSEQ);
  cvt(ctx, ctx_bf, MROWS * DMODEL);
  gemmF(ctx_bf, saWo_bf, sa_bo, proj, MROWS, DMODEL, DMODEL);
  ln_kernel<<<MROWS, 256, 0, stream>>>(x, proj, ln1_g, ln1_b, yres, y_bf);

  // --- cross attention ---
  gemmB(y_bf,   caWq_bf, ca_bq, Qb, MROWS, DMODEL, DMODEL);
  gemmB(enc_bf, caWk_bf, ca_bk, Kb, MROWS, DMODEL, DMODEL);
  gemmB(enc_bf, caWv_bf, ca_bv, Vb, MROWS, DMODEL, DMODEL);
  transpose_v_kernel<<<dim3(SSEQ / 32, NHEAD, NB), 256, 0, stream>>>(Vb, Vt, SSEQ);
  attn_kernel<<<dim3(TSEQ / 64, NHEAD, NB), 128, 0, stream>>>(
      Qb, Kb, Vt, src_mask, ctx, TSEQ, SSEQ);
  cvt(ctx, ctx_bf, MROWS * DMODEL);
  gemmF(ctx_bf, caWo_bf, ca_bo, proj, MROWS, DMODEL, DMODEL);
  ln_kernel<<<MROWS, 256, 0, stream>>>(yres, proj, ln2_g, ln2_b, yres, y_bf);

  // --- feed forward ---
  gemm_kernel<true, true><<<dim3(MROWS / 128, DFFN / 128), 256, 0, stream>>>(
      y_bf, W1_bf, ff_b1, (void*)hidden, MROWS, DFFN, DMODEL);
  gemmF(hidden, W2_bf, ff_b2, proj, MROWS, DMODEL, DFFN);
  ln_kernel<<<MROWS, 256, 0, stream>>>(yres, proj, ln3_g, ln3_b, (float*)d_out, y_bf);
}
